// Attention_74302934221121
// MI455X (gfx1250) — matrix operand order repaired and checked by reading, not compiled, not run on hardware
//
#include <hip/hip_runtime.h>

// CDNA5 / gfx1250 wave32 attention.
// Fast path: bf16x3 split-precision WMMA (v_wmma_f32_16x16x32_bf16) for the
// logit GEMM, plain bf16 WMMA for the context GEMM. Needs 96MB of d_ws for
// pre-converted V (hi/lo row-major + hi transposed).
// Fallback:  full-f32 WMMA (v_wmma_f32_16x16x4_f32) if d_ws is too small.

typedef __attribute__((ext_vector_type(2)))  float  v2f;
typedef __attribute__((ext_vector_type(4)))  float  v4f;
typedef __attribute__((ext_vector_type(8)))  float  v8f;
typedef __bf16 bf16;
typedef __attribute__((ext_vector_type(4)))  __bf16 v4bf;
typedef __attribute__((ext_vector_type(8)))  __bf16 v8bf;
typedef __attribute__((ext_vector_type(16))) __bf16 v16bf;

#define BATCH 8
#define SEQL  2048
#define DIM   1024

static __device__ __forceinline__ v16bf join8(v8bf a, v8bf b) {
    return __builtin_shufflevector(a, b, 0,1,2,3,4,5,6,7,8,9,10,11,12,13,14,15);
}

// ------------------------------------------------------------------
// Pass 0a: f32 -> (bf16 hi, bf16 lo) row-major copies of V.
// ------------------------------------------------------------------
__global__ __launch_bounds__(256) void
cvt_hilo_kernel(const float* __restrict__ src, bf16* __restrict__ hi,
                bf16* __restrict__ lo) {
    size_t i = ((size_t)blockIdx.x * 256 + threadIdx.x) * 4;
    v4f x = *(const v4f*)(src + i);
    v4bf h, l;
#pragma unroll
    for (int j = 0; j < 4; ++j) {
        h[j] = (bf16)x[j];
        l[j] = (bf16)(x[j] - (float)h[j]);
    }
    *(v4bf*)(hi + i) = h;
    *(v4bf*)(lo + i) = l;
}

// ------------------------------------------------------------------
// Pass 0b: Vt[b][d][n] = bf16(V[b][n][d])  (32x32 LDS tile transpose)
// ------------------------------------------------------------------
__global__ __launch_bounds__(256) void
transpose_hi_kernel(const float* __restrict__ src, bf16* __restrict__ dst) {
    __shared__ bf16 t[32][33];
    const int blk = blockIdx.x;
    const int b   = blk >> 11;            // / (32 ktiles * 64 ntiles)
    const int r   = blk & 2047;
    const int k0  = (r >> 6) << 5;        // 32 k-tiles
    const int n0  = (r & 63) << 5;        // 64 n-tiles
    const int tx  = threadIdx.x & 31;
    const int ty  = threadIdx.x >> 5;     // 0..7
#pragma unroll
    for (int it = 0; it < 4; ++it) {
        const int n = n0 + ty + it * 8;
        t[ty + it * 8][tx] = (bf16)src[((size_t)b * SEQL + n) * DIM + k0 + tx];
    }
    __syncthreads();
#pragma unroll
    for (int it = 0; it < 4; ++it) {
        const int k = k0 + ty + it * 8;
        dst[((size_t)b * DIM + k) * SEQL + n0 + tx] = t[tx][ty + it * 8];
    }
}

// ------------------------------------------------------------------
// Main fast-path kernel: bf16x3 logits + softmax + bf16 context.
// One WG = one batch x 16 query rows, 128 threads (4 wave32s).
// ------------------------------------------------------------------
__global__ __launch_bounds__(128) void
attn_bf16_kernel(const float* __restrict__ Q,
                 const bf16* __restrict__ Vhi,
                 const bf16* __restrict__ Vlo,
                 const bf16* __restrict__ Vt,
                 float* __restrict__ ctx,
                 float* __restrict__ W) {
    const int wg       = blockIdx.x;
    const int b        = wg >> 7;
    const int qt       = wg & 127;
    const int row_base = qt << 4;

    const int tid  = threadIdx.x;
    const int wave = tid >> 5;
    const int lane = tid & 31;
    const int mn   = lane & 15;   // M (A frag) / N (B frag)
    const int h    = lane >> 4;   // lane-half K split per ISA layout

    const float* Qb = Q   + (size_t)b * SEQL * DIM;
    float*       Wb = W   + (size_t)b * SEQL * SEQL;
    float*       Cb = ctx + (size_t)b * SEQL * DIM;

    // 64KB: Q hi/lo during phase A, reused as bf16 P (16x2048) in phase B2.
    __shared__ bf16 s_q[2][16 * DIM];
    bf16* s_p = &s_q[0][0];
    __shared__ float s_red[16][8];
    __shared__ float s_m[16];
    __shared__ float s_invl[16];

    // ---- Stage Q block as bf16 hi/lo into LDS (split once, reused 128x) ----
    {
        const int r  = tid >> 3;
        const int c0 = (tid & 7) << 7;     // 128 cols per thread
        const float* qsrc = Qb + (size_t)(row_base + r) * DIM + c0;
        for (int c = 0; c < 128; c += 4) {
            v4f x = *(const v4f*)(qsrc + c);
            v4bf hh, ll;
#pragma unroll
            for (int j = 0; j < 4; ++j) {
                hh[j] = (bf16)x[j];
                ll[j] = (bf16)(x[j] - (float)hh[j]);
            }
            *(v4bf*)(&s_q[0][r * DIM + c0 + c]) = hh;
            *(v4bf*)(&s_q[1][r * DIM + c0 + c]) = ll;
        }
    }
    __syncthreads();

    // ---- Phase A: raw scores S = Q @ V^T via bf16x3 WMMA ----
    // A frag (16x32 bf16): lane(mn,h) -> Q[mn][k+8h..+7] ++ Q[mn][k+16+8h..+7]
    // B frag (32x16 bf16): lane(mn,h) -> V[col+mn][k+8h..+7] ++ V[col+mn][k+16+8h..+7]
    for (int ct = 0; ct < 32; ++ct) {
        const int col_base = (((ct << 2) + wave) << 4);
        const bf16* bhp = Vhi + ((size_t)b * SEQL + col_base + mn) * DIM + 8 * h;
        const bf16* blp = Vlo + ((size_t)b * SEQL + col_base + mn) * DIM + 8 * h;
        v8f acc = {};
        for (int k = 0; k < DIM; k += 32) {
            v16bf ahi = join8(*(const v8bf*)(&s_q[0][mn * DIM + k +      8 * h]),
                              *(const v8bf*)(&s_q[0][mn * DIM + k + 16 + 8 * h]));
            v16bf alo = join8(*(const v8bf*)(&s_q[1][mn * DIM + k +      8 * h]),
                              *(const v8bf*)(&s_q[1][mn * DIM + k + 16 + 8 * h]));
            v16bf bhi = join8(*(const v8bf*)(bhp + k), *(const v8bf*)(bhp + k + 16));
            v16bf blo = join8(*(const v8bf*)(blp + k), *(const v8bf*)(blp + k + 16));
            acc = __builtin_amdgcn_wmma_f32_16x16x32_bf16(false, ahi, false, bhi,
                                                          (short)0, acc, false, false);
            acc = __builtin_amdgcn_wmma_f32_16x16x32_bf16(false, ahi, false, blo,
                                                          (short)0, acc, false, false);
            acc = __builtin_amdgcn_wmma_f32_16x16x32_bf16(false, alo, false, bhi,
                                                          (short)0, acc, false, false);
        }
        float* wout = Wb + (size_t)(row_base + 8 * h) * SEQL + col_base + mn;
#pragma unroll
        for (int r = 0; r < 8; ++r)
            wout[(size_t)r * SEQL] = acc[r];
    }
    __threadfence();
    __syncthreads();

    // ---- Phase B1: row softmax in place; also stash bf16 P in LDS ----
    const int rr   = tid >> 3;
    const int cseg = (tid & 7) << 8;
    float* srow = Wb + (size_t)(row_base + rr) * SEQL + cseg;

    float pm = -3.402823466e38f;
    for (int c = 0; c < 256; c += 4) {
        v4f x = *(const v4f*)(srow + c);
        pm = fmaxf(pm, fmaxf(fmaxf(x[0], x[1]), fmaxf(x[2], x[3])));
    }
    s_red[rr][tid & 7] = pm;
    __syncthreads();
    if (tid < 16) {
        float m = s_red[tid][0];
#pragma unroll
        for (int i = 1; i < 8; ++i) m = fmaxf(m, s_red[tid][i]);
        s_m[tid] = m;
    }
    __syncthreads();
    const float mrow = s_m[rr];

    float ps = 0.f;
    for (int c = 0; c < 256; c += 4) {
        v4f x = *(const v4f*)(srow + c);
        ps += __expf(x[0] - mrow) + __expf(x[1] - mrow) +
              __expf(x[2] - mrow) + __expf(x[3] - mrow);
    }
    s_red[rr][tid & 7] = ps;
    __syncthreads();
    if (tid < 16) {
        float s = 0.f;
#pragma unroll
        for (int i = 0; i < 8; ++i) s += s_red[tid][i];
        s_invl[tid] = 1.0f / s;
    }
    __syncthreads();
    const float invl = s_invl[rr];

    for (int c = 0; c < 256; c += 4) {
        v4f x = *(const v4f*)(srow + c);
        v4f y;
        v4bf pb;
#pragma unroll
        for (int j = 0; j < 4; ++j) {
            y[j]  = __expf(x[j] - mrow) * invl;
            pb[j] = (bf16)y[j];
        }
        *(v4f*)(srow + c) = y;                          // final weights (f32)
        *(v4bf*)(&s_p[rr * SEQL + cseg + c]) = pb;      // bf16 P for phase B2
    }
    __syncthreads();

    // ---- Phase B2: context = P @ V (plain bf16 WMMA) ----
    // A frag from LDS P; B frag from transposed Vt (contiguous per lane).
    for (int dt = 0; dt < 16; ++dt) {
        const int dcol = (wave << 8) + (dt << 4);
        const bf16* bp = Vt + ((size_t)b * DIM + dcol + mn) * SEQL + 8 * h;
        v8f acc = {};
        for (int kv = 0; kv < SEQL; kv += 32) {
            v16bf a = join8(*(const v8bf*)(&s_p[mn * SEQL + kv +      8 * h]),
                            *(const v8bf*)(&s_p[mn * SEQL + kv + 16 + 8 * h]));
            v16bf bb = join8(*(const v8bf*)(bp + kv), *(const v8bf*)(bp + kv + 16));
            acc = __builtin_amdgcn_wmma_f32_16x16x32_bf16(false, a, false, bb,
                                                          (short)0, acc, false, false);
        }
        float* cout = Cb + (size_t)(row_base + 8 * h) * DIM + dcol + mn;
#pragma unroll
        for (int r = 0; r < 8; ++r)
            cout[(size_t)r * DIM] = acc[r];
    }
}

// ------------------------------------------------------------------
// Fallback: full-f32 WMMA version (no workspace needed).
// ------------------------------------------------------------------
__global__ __launch_bounds__(128) void
attn_f32_kernel(const float* __restrict__ Q, const float* __restrict__ V,
                float* __restrict__ ctx, float* __restrict__ W) {
    const int wg       = blockIdx.x;
    const int b        = wg >> 7;
    const int qt       = wg & 127;
    const int row_base = qt << 4;

    const int tid  = threadIdx.x;
    const int wave = tid >> 5;
    const int lane = tid & 31;
    const int mn   = lane & 15;
    const int h    = lane >> 4;

    const float* Qb = Q   + (size_t)b * SEQL * DIM;
    const float* Vb = V   + (size_t)b * SEQL * DIM;
    float*       Wb = W   + (size_t)b * SEQL * SEQL;
    float*       Cb = ctx + (size_t)b * SEQL * DIM;

    __shared__ float s_red[16][8];
    __shared__ float s_m[16];
    __shared__ float s_invl[16];

    const float* qrow = Qb + (size_t)(row_base + mn) * DIM + 2 * h;
    for (int ct = 0; ct < 32; ++ct) {
        const int col_base = (((ct << 2) + wave) << 4);
        const float* vrow  = Vb + (size_t)(col_base + mn) * DIM + 2 * h;
        v8f acc = {};
#pragma unroll 4
        for (int k = 0; k < DIM; k += 4) {
            v2f a  = *(const v2f*)(qrow + k);
            v2f bf = *(const v2f*)(vrow + k);
            acc = __builtin_amdgcn_wmma_f32_16x16x4_f32(
                false, a, false, bf, (short)0, acc, false, false);
        }
        float* wout = Wb + (size_t)(row_base + 8 * h) * SEQL + col_base + mn;
#pragma unroll
        for (int r = 0; r < 8; ++r)
            wout[(size_t)r * SEQL] = acc[r];
    }
    __threadfence();
    __syncthreads();

    const int rr   = tid >> 3;
    const int cseg = (tid & 7) << 8;
    float* srow = Wb + (size_t)(row_base + rr) * SEQL + cseg;

    float pm = -3.402823466e38f;
    for (int c = 0; c < 256; c += 4) {
        v4f x = *(const v4f*)(srow + c);
        pm = fmaxf(pm, fmaxf(fmaxf(x[0], x[1]), fmaxf(x[2], x[3])));
    }
    s_red[rr][tid & 7] = pm;
    __syncthreads();
    if (tid < 16) {
        float m = s_red[tid][0];
#pragma unroll
        for (int i = 1; i < 8; ++i) m = fmaxf(m, s_red[tid][i]);
        s_m[tid] = m;
    }
    __syncthreads();
    const float mrow = s_m[rr];

    float ps = 0.f;
    for (int c = 0; c < 256; c += 4) {
        v4f x = *(const v4f*)(srow + c);
        ps += __expf(x[0] - mrow) + __expf(x[1] - mrow) +
              __expf(x[2] - mrow) + __expf(x[3] - mrow);
    }
    s_red[rr][tid & 7] = ps;
    __syncthreads();
    if (tid < 16) {
        float s = 0.f;
#pragma unroll
        for (int i = 0; i < 8; ++i) s += s_red[tid][i];
        s_invl[tid] = 1.0f / s;
    }
    __syncthreads();
    const float invl = s_invl[rr];

    for (int c = 0; c < 256; c += 4) {
        v4f x = *(const v4f*)(srow + c);
        v4f y;
#pragma unroll
        for (int j = 0; j < 4; ++j) y[j] = __expf(x[j] - mrow) * invl;
        *(v4f*)(srow + c) = y;
    }
    __threadfence();
    __syncthreads();

    const float* prow = Wb + (size_t)(row_base + mn) * SEQL + 2 * h;
    for (int dt = 0; dt < 16; ++dt) {
        const int dcol = (wave << 8) + (dt << 4);
        const float* vcol = Vb + dcol + mn + (size_t)(2 * h) * DIM;
        v8f acc = {};
#pragma unroll 4
        for (int kv = 0; kv < SEQL; kv += 4) {
            v2f a = *(const v2f*)(prow + kv);
            v2f bf;
            bf[0] = vcol[(size_t)kv * DIM];
            bf[1] = vcol[(size_t)kv * DIM + DIM];
            acc = __builtin_amdgcn_wmma_f32_16x16x4_f32(
                false, a, false, bf, (short)0, acc, false, false);
        }
        float* cout = Cb + (size_t)(row_base + 8 * h) * DIM + dcol + mn;
#pragma unroll
        for (int r = 0; r < 8; ++r)
            cout[(size_t)r * DIM] = acc[r];
    }
}

extern "C" void kernel_launch(void* const* d_in, const int* in_sizes, int n_in,
                              void* d_out, int out_size, void* d_ws, size_t ws_size,
                              hipStream_t stream) {
    const float* Q = (const float*)d_in[0];
    const float* V = (const float*)d_in[1];
    float* ctx = (float*)d_out;                                   // [8,2048,1024]
    float* W   = (float*)d_out + (size_t)BATCH * SEQL * DIM;      // [8,2048,2048]

    const size_t nElem = (size_t)BATCH * SEQL * DIM;              // 16.8M
    const size_t need  = 3 * nElem * sizeof(bf16);                // Vhi+Vlo+Vt = 96MB

    if (ws_size >= need) {
        bf16* Vhi = (bf16*)d_ws;
        bf16* Vlo = Vhi + nElem;
        bf16* Vt  = Vlo + nElem;
        cvt_hilo_kernel<<<dim3(nElem / 1024), dim3(256), 0, stream>>>(V, Vhi, Vlo);
        transpose_hi_kernel<<<dim3(BATCH * (DIM / 32) * (SEQL / 32)), dim3(256), 0,
                              stream>>>(V, Vt);
        attn_bf16_kernel<<<dim3(BATCH * (SEQL / 16)), dim3(128), 0, stream>>>(
            Q, Vhi, Vlo, Vt, ctx, W);
    } else {
        attn_f32_kernel<<<dim3(BATCH * (SEQL / 16)), dim3(128), 0, stream>>>(
            Q, V, ctx, W);
    }
}
